// SelfAttention_55748675502748
// MI455X (gfx1250) — matrix operand order repaired and checked by reading, not compiled, not run on hardware
//
#include <hip/hip_runtime.h>

// ---------------------------------------------------------------------------
// Types for CDNA5 WMMA (wave32): A/B = 16 x bf16 per lane, C/D = 8 x f32.
// ---------------------------------------------------------------------------
typedef __attribute__((ext_vector_type(16))) __bf16 v16bf;
typedef __attribute__((ext_vector_type(8)))  float  v8f;
typedef __attribute__((ext_vector_type(4)))  unsigned int u32x4;
typedef __attribute__((ext_vector_type(4)))  float  f32x4;

union Frag16 { v16bf v; u32x4 q[2]; };

__device__ __forceinline__ __bf16 f2bf(float f) {
  union { float f; unsigned u; } v; v.f = f;
  unsigned r = v.u + 0x7FFFu + ((v.u >> 16) & 1u);   // round-to-nearest-even
  unsigned short h = (unsigned short)(r >> 16);
  return __builtin_bit_cast(__bf16, h);
}

// Per-lane async copy global -> LDS (16B), tracked by ASYNCcnt.
// VDST = per-lane LDS byte address (wave-relative: low 32 bits of the flat
// shared-aperture address, ISA 10.2), VADDR = 64-bit global address.
__device__ __forceinline__ void async_ld_b128(unsigned ldsOff, const void* g) {
  asm volatile("global_load_async_to_lds_b128 %0, %1, off"
               :: "v"(ldsOff), "v"(g) : "memory");
}
__device__ __forceinline__ void wait_async0() {
  asm volatile("s_wait_asynccnt 0x0" ::: "memory");
}

// ---------------------------------------------------------------------------
// Elementwise f32 -> bf16 (vectorized x4)
// ---------------------------------------------------------------------------
__global__ void cvt_bf16_kernel(const float* __restrict__ in,
                                __bf16* __restrict__ out, long long n4) {
  long long i = (long long)blockIdx.x * blockDim.x + threadIdx.x;
  if (i >= n4) return;
  f32x4 v = ((const f32x4* __restrict__)in)[i];
  union { __bf16 h[4]; unsigned long long u; } o;
#pragma unroll
  for (int r = 0; r < 4; ++r) o.h[r] = f2bf(v[r]);
  ((unsigned long long* __restrict__)out)[i] = o.u;
}

// f32 (K x N) -> bf16 transposed (N x K): B operand is always N-major.
__global__ void cvtT_bf16_kernel(const float* __restrict__ W,
                                 __bf16* __restrict__ Wt, int D) {
  int idx = blockIdx.x * 256 + threadIdx.x;
  int k = idx / D, n = idx % D;
  Wt[(size_t)n * D + k] = f2bf(W[(size_t)k * D + n]);
}

// ---------------------------------------------------------------------------
// Unified WMMA GEMM:  C[M,N] = A[M,K] (row-major, bf16) * B[N,K]^T (N-major bf16)
// Block = 256 threads (8 waves) -> 128x128 tile, BK = 64 (two wmma K-steps).
// Tiles are fed with GLOBAL_LOAD_ASYNC_TO_LDS_B128 (ASYNCcnt) into a
// double-buffered LDS: tile t+1's asyncs are issued before the 16-WMMA block
// of tile t, drained with s_wait_asynccnt 0 afterwards, one barrier per tile.
// Tile t+2 is prefetched (global_prefetch_b8). No VGPR staging, no ds_store.
// Wave grid 4(M) x 2(N); each wave: 2x4 frags of 16x16 wmma_f32_16x16x32_bf16.
// OUT_BF16: store bf16 (QKV proj). OUT_TRANS: store C^T (for V -> V^T).
// skipUpper: drop blocks entirely above the causal diagonal (scores pass).
// ---------------------------------------------------------------------------
template <bool OUT_BF16, bool OUT_TRANS>
__global__ __launch_bounds__(256) void gemm_wmma_kernel(
    const __bf16* __restrict__ A, const __bf16* __restrict__ Bmat,
    void* __restrict__ Cout,
    int K, int lda, int ldb, int ldc,
    long long sA, long long sB, long long sC, int skipUpper) {
  const int bz = blockIdx.z;
  const __bf16* __restrict__ Ab = A + (size_t)bz * sA;
  const __bf16* __restrict__ Bb = Bmat + (size_t)bz * sB;
  const int mBase = blockIdx.y * 128;
  const int nBase = blockIdx.x * 128;
  if (skipUpper && nBase > mBase + 127) return;   // block-uniform early out

  // Row = 64 bf16 padded to 9 u32x4 (144B): 16B aligned, and start banks are
  // distinct across any 16 consecutive rows (36*dr % 64 == 0 iff dr % 16 == 0).
  __shared__ u32x4 As[2][128][9];
  __shared__ u32x4 Bs[2][128][9];

  const int tid  = threadIdx.x;
  const int lane = tid & 31;
  const int wid  = tid >> 5;
  const int lrow = lane & 15;   // M (A) / N (B,D) index within 16
  const int lg   = lane >> 4;   // half-wave group
  const int waveM = wid & 3;    // 4 wave-rows  * 32 rows
  const int waveN = wid >> 2;   // 2 wave-cols  * 64 cols

  v8f acc[2][4];
#pragma unroll
  for (int mi = 0; mi < 2; ++mi)
#pragma unroll
    for (int ni = 0; ni < 4; ++ni)
      acc[mi][ni] = v8f{0.f, 0.f, 0.f, 0.f, 0.f, 0.f, 0.f, 0.f};

  // Tile feed: thread t owns a 64B run (4 x async b128) of one row per tile.
  const int lr  = tid >> 1;           // 0..127 : tile row
  const int lq4 = (tid & 1) * 4;      // u32x4 index base (k 0..31 | 32..63)
  const u32x4* __restrict__ Ag =
      (const u32x4*)(Ab + (size_t)(mBase + lr) * lda) + lq4;
  const u32x4* __restrict__ Bg =
      (const u32x4*)(Bb + (size_t)(nBase + lr) * ldb) + lq4;

  // Per-buffer LDS byte addresses of this thread's store slot.
  const unsigned ldsA[2] = {(unsigned)(uintptr_t)&As[0][lr][lq4],
                            (unsigned)(uintptr_t)&As[1][lr][lq4]};
  const unsigned ldsB[2] = {(unsigned)(uintptr_t)&Bs[0][lr][lq4],
                            (unsigned)(uintptr_t)&Bs[1][lr][lq4]};

  const int ntiles = K >> 6;          // K multiple of 64 in all passes

  auto issue_tile = [&](int buf) {
#pragma unroll
    for (int i = 0; i < 4; ++i) {
      async_ld_b128(ldsA[buf] + 16u * i, (const void*)(Ag + i));
      async_ld_b128(ldsB[buf] + 16u * i, (const void*)(Bg + i));
    }
    Ag += 8; Bg += 8;
  };

  // Prologue: tile 0 -> buffer 0.
  issue_tile(0);
  wait_async0();
  __syncthreads();

  for (int t = 0; t < ntiles; ++t) {
    const int cur = t & 1;
    const bool more = (t + 1 < ntiles);
    if (more) {
      issue_tile(cur ^ 1);              // Ag/Bg now point at tile t+2
      if (t + 2 < ntiles) {
        __builtin_prefetch((const void*)Ag, 0, 1);   // global_prefetch_b8
        __builtin_prefetch((const void*)Bg, 0, 1);
      }
    }

#pragma unroll
    for (int kk2 = 0; kk2 < 2; ++kk2) {          // k offsets 0, 32
      Frag16 af[2], bfr[4];
#pragma unroll
      for (int mi = 0; mi < 2; ++mi) {
        int row = waveM * 32 + mi * 16 + lrow;
        // A 16x32 bf16 layout: group g holds K = g*8+[0,8) and 16+g*8+[0,8)
        af[mi].q[0] = As[cur][row][kk2 * 4 + lg];
        af[mi].q[1] = As[cur][row][kk2 * 4 + 2 + lg];
      }
#pragma unroll
      for (int ni = 0; ni < 4; ++ni) {
        int row = waveN * 64 + ni * 16 + lrow;
        // B 32x16 bf16 layout: group g holds K = g*8+[0,8) and 16+g*8+[0,8)
        bfr[ni].q[0] = Bs[cur][row][kk2 * 4 + lg];
        bfr[ni].q[1] = Bs[cur][row][kk2 * 4 + 2 + lg];
      }
#pragma unroll
      for (int mi = 0; mi < 2; ++mi)
#pragma unroll
        for (int ni = 0; ni < 4; ++ni)
          acc[mi][ni] = __builtin_amdgcn_wmma_f32_16x16x32_bf16(
              false, af[mi].v, false, bfr[ni].v, (short)0, acc[mi][ni],
              false, false);
    }

    if (more) wait_async0();            // drain tile t+1's async copies
    __syncthreads();
  }

  // Epilogue. D layout: lane holds N = lrow, rows M = 8*lg + r (r = VGPR idx).
  const int mW = mBase + waveM * 32;
  const int nW = nBase + waveN * 64;
  if (OUT_TRANS) {
    __bf16* __restrict__ Ct = (__bf16*)Cout + (size_t)bz * sC;
#pragma unroll
    for (int mi = 0; mi < 2; ++mi)
#pragma unroll
      for (int ni = 0; ni < 4; ++ni) {
        int n  = nW + ni * 16 + lrow;
        int m0 = mW + mi * 16 + 8 * lg;
        union { u32x4 q; __bf16 h[8]; } o;
#pragma unroll
        for (int r = 0; r < 8; ++r) o.h[r] = f2bf(acc[mi][ni][r]);
        *(u32x4*)(Ct + (size_t)n * ldc + m0) = o.q;   // 16B contiguous in m
      }
  } else if (OUT_BF16) {
    __bf16* __restrict__ Cb = (__bf16*)Cout + (size_t)bz * sC;
#pragma unroll
    for (int mi = 0; mi < 2; ++mi)
#pragma unroll
      for (int ni = 0; ni < 4; ++ni) {
        int n = nW + ni * 16 + lrow;
#pragma unroll
        for (int r = 0; r < 8; ++r) {
          int m = mW + mi * 16 + 8 * lg + r;
          Cb[(size_t)m * ldc + n] = f2bf(acc[mi][ni][r]);
        }
      }
  } else {
    float* __restrict__ Cf = (float*)Cout + (size_t)bz * sC;
#pragma unroll
    for (int mi = 0; mi < 2; ++mi)
#pragma unroll
      for (int ni = 0; ni < 4; ++ni) {
        int n = nW + ni * 16 + lrow;
#pragma unroll
        for (int r = 0; r < 8; ++r) {
          int m = mW + mi * 16 + 8 * lg + r;
          Cf[(size_t)m * ldc + n] = acc[mi][ni][r];
        }
      }
  }
}

// ---------------------------------------------------------------------------
// Causal row softmax: P[b,i,j] = softmax_j<=i(S[b,i,j]/32), bf16 out, upper
// triangle zeroed so P@V can run as a dense GEMM.
// ---------------------------------------------------------------------------
__global__ __launch_bounds__(256) void softmax_causal_kernel(
    const float* __restrict__ S, __bf16* __restrict__ P, int C) {
  const int row = blockIdx.x, b = blockIdx.y, tid = threadIdx.x;
  const float* __restrict__ s = S + ((size_t)b * C + row) * (size_t)C;
  __bf16* __restrict__ p = P + ((size_t)b * C + row) * (size_t)C;
  const int n = row + 1;                       // valid causal length
  __shared__ float red[256];

  float m = -3.4e38f;
  for (int j = tid; j < n; j += 256) m = fmaxf(m, s[j]);
  red[tid] = m; __syncthreads();
  for (int o = 128; o > 0; o >>= 1) {
    if (tid < o) red[tid] = fmaxf(red[tid], red[tid + o]);
    __syncthreads();
  }
  m = red[0]; __syncthreads();

  const float c = 0.04508422f;                 // (1/32) * log2(e)
  float sum = 0.f;
  for (int j = tid; j < n; j += 256) sum += exp2f((s[j] - m) * c);
  red[tid] = sum; __syncthreads();
  for (int o = 128; o > 0; o >>= 1) {
    if (tid < o) red[tid] += red[tid + o];
    __syncthreads();
  }
  const float inv = 1.0f / red[0];

  for (int j = tid; j < n; j += 256) p[j] = f2bf(exp2f((s[j] - m) * c) * inv);
  for (int j = n + tid; j < C; j += 256) p[j] = f2bf(0.0f);
}

// ---------------------------------------------------------------------------
// Driver
// ---------------------------------------------------------------------------
extern "C" void kernel_launch(void* const* d_in, const int* in_sizes, int n_in,
                              void* d_out, int out_size, void* d_ws,
                              size_t ws_size, hipStream_t stream) {
  (void)in_sizes; (void)n_in; (void)out_size; (void)ws_size;
  constexpr long long Bn = 4, Cn = 2048, Dn = 1024;

  const float* x  = (const float*)d_in[0];
  const float* Wq = (const float*)d_in[1];
  const float* Wk = (const float*)d_in[2];
  const float* Wv = (const float*)d_in[3];

  char* ws = (char*)d_ws;
  size_t off = 0;
  auto take = [&](size_t bytes) -> char* {
    char* p = ws + off;
    off += (bytes + 255) & ~(size_t)255;
    return p;
  };
  __bf16* xb  = (__bf16*)take(Bn * Cn * Dn * 2);
  __bf16* WqT = (__bf16*)take(Dn * Dn * 2);
  __bf16* WkT = (__bf16*)take(Dn * Dn * 2);
  __bf16* WvT = (__bf16*)take(Dn * Dn * 2);
  __bf16* Qb  = (__bf16*)take(Bn * Cn * Dn * 2);
  __bf16* Kb  = (__bf16*)take(Bn * Cn * Dn * 2);
  __bf16* VTb = (__bf16*)take(Bn * Cn * Dn * 2);   // per batch: (Dn x Cn)
  float*  Sf  = (float*) take(Bn * Cn * Cn * 4);
  __bf16* Pb  = (__bf16*)take(Bn * Cn * Cn * 2);

  // 1) Convert inputs to bf16 (weights also transposed -> N-major B operand).
  {
    long long n4 = Bn * Cn * Dn / 4;
    cvt_bf16_kernel<<<dim3((unsigned)(n4 / 256)), 256, 0, stream>>>(x, xb, n4);
    unsigned g = (unsigned)(Dn * Dn / 256);
    cvtT_bf16_kernel<<<g, 256, 0, stream>>>(Wq, WqT, (int)Dn);
    cvtT_bf16_kernel<<<g, 256, 0, stream>>>(Wk, WkT, (int)Dn);
    cvtT_bf16_kernel<<<g, 256, 0, stream>>>(Wv, WvT, (int)Dn);
  }

  const long long sXC = Cn * Dn;     // per-batch x/Q/K stride
  const long long sVT = Dn * Cn;     // per-batch V^T stride
  const long long sS  = Cn * Cn;     // per-batch scores / P stride

  // 2) QKV projections (bf16 out). Grid: N/128 x M/128 x batch.
  dim3 gProj((unsigned)(Dn / 128), (unsigned)(Cn / 128), (unsigned)Bn);
  gemm_wmma_kernel<true, false><<<gProj, 256, 0, stream>>>(
      xb, WqT, Qb, (int)Dn, (int)Dn, (int)Dn, (int)Dn, sXC, 0, sXC, 0);
  gemm_wmma_kernel<true, false><<<gProj, 256, 0, stream>>>(
      xb, WkT, Kb, (int)Dn, (int)Dn, (int)Dn, (int)Dn, sXC, 0, sXC, 0);
  gemm_wmma_kernel<true, true><<<gProj, 256, 0, stream>>>(
      xb, WvT, VTb, (int)Dn, (int)Dn, (int)Dn, (int)Cn, sXC, 0, sVT, 0);

  // 3) Scores S = Q K^T (f32), lower-triangular blocks only.
  dim3 gScore((unsigned)(Cn / 128), (unsigned)(Cn / 128), (unsigned)Bn);
  gemm_wmma_kernel<false, false><<<gScore, 256, 0, stream>>>(
      Qb, Kb, Sf, (int)Dn, (int)Dn, (int)Dn, (int)Cn, sXC, sXC, sS, 1);

  // 4) Causal softmax -> P (bf16, upper triangle zeroed).
  softmax_causal_kernel<<<dim3((unsigned)Cn, (unsigned)Bn), 256, 0, stream>>>(
      Sf, Pb, (int)Cn);

  // 5) O = P V  (A = P [CnxCn], B = V^T N-major [Dn x Cn]) -> f32 d_out.
  dim3 gOut((unsigned)(Dn / 128), (unsigned)(Cn / 128), (unsigned)Bn);
  gemm_wmma_kernel<false, false><<<gOut, 256, 0, stream>>>(
      Pb, VTb, (float*)d_out, (int)Cn, (int)Cn, (int)Cn, (int)Dn,
      sS, sVT, sXC, 0);
}
